// RPNPostProcessor_13615046329061
// MI455X (gfx1250) — compile-verified, matrix operand order checked by reading, not run on hardware
//
#include <hip/hip_runtime.h>

// ---------------- problem constants ----------------
#define NB     2                       // batches
#define AA     3                       // anchors per cell
#define SP     (64*128*128)            // spatial cells per batch = 1048576
#define FLATN  (SP*AA)                 // 3145728 flat anchors per batch
#define TOTE   (NB*AA*SP)              // total objectness elements
#define PRE    2000
#define POST   1000
#define CAP    8192                    // candidate buffer per batch
#define NBIN   2048                    // radix histogram bins (top 11 key bits)
#define WORDS  63                      // ceil(PRE/32) mask words per row
#define RTILES 125                     // PRE/16 row tiles
#define CTILES 63                      // col word tiles (32 cols each)
#define NEGV   (-1e30f)
#define CLIPV  4.135166556742356f      // log(1000/16)
#define VSCALE (1.0f/16384.0f)         // keep volumes inside f16 range
#define VSCALI 16384.0f

typedef __attribute__((ext_vector_type(16))) _Float16 v16h;
typedef __attribute__((ext_vector_type(8)))  float    v8f;

// ---------------- workspace layout (32-bit words) ----------------
constexpr size_t OFF_HIST = 0;                         // NB*NBIN
constexpr size_t OFF_CNT  = OFF_HIST + (size_t)NB*NBIN;// NB
constexpr size_t OFF_THR  = OFF_CNT  + NB;             // NB
constexpr size_t OFF_CVAL = OFF_THR  + NB;             // NB*CAP (float)
constexpr size_t OFF_CIDX = OFF_CVAL + (size_t)NB*CAP; // NB*CAP
constexpr size_t OFF_TIDX = OFF_CIDX + (size_t)NB*CAP; // NB*PRE
constexpr size_t OFF_TRAW = OFF_TIDX + (size_t)NB*PRE; // NB*PRE (float)
constexpr size_t OFF_TSC  = OFF_TRAW + (size_t)NB*PRE; // NB*PRE (float)
constexpr size_t OFF_PROP = OFF_TSC  + (size_t)NB*PRE; // NB*PRE*6 (float)
constexpr size_t OFF_VAL  = OFF_PROP + (size_t)NB*PRE*6;// NB*PRE
constexpr size_t OFF_SC2  = OFF_VAL  + (size_t)NB*PRE; // NB*PRE (float)
constexpr size_t OFF_MASK = OFF_SC2  + (size_t)NB*PRE; // NB*PRE*WORDS

// ---------------- helpers ----------------
__device__ __forceinline__ unsigned fkey(float x) {
  unsigned u = __float_as_uint(x);
  return (u & 0x80000000u) ? ~u : (u | 0x80000000u);   // order-preserving key
}
__device__ __forceinline__ float ikey(unsigned k) {
  unsigned u = (k & 0x80000000u) ? (k ^ 0x80000000u) : ~k;
  return __uint_as_float(u);
}
__device__ __forceinline__ unsigned ballot32(bool p) {
#if __has_builtin(__builtin_amdgcn_ballot_w32)
  return __builtin_amdgcn_ballot_w32(p);
#else
  return (unsigned)__ballot(p);
#endif
}

// descending bitonic sort of nn (power of two) u64 keys in LDS
__device__ __forceinline__ void bitonic_desc(unsigned long long* s, int nn) {
  for (int k = 2; k <= nn; k <<= 1) {
    for (int j = k >> 1; j > 0; j >>= 1) {
      __syncthreads();
      for (int t = threadIdx.x; t < nn; t += blockDim.x) {
        int ixj = t ^ j;
        if (ixj > t) {
          unsigned long long a = s[t], b = s[ixj];
          bool descSeg = ((t & k) == 0);
          if (descSeg ? (a < b) : (a > b)) { s[t] = b; s[ixj] = a; }
        }
      }
    }
  }
  __syncthreads();
}

// ---------------- K0: zero histogram + counters ----------------
__global__ void k0_zero(unsigned* ws) {
  int t = blockIdx.x * blockDim.x + threadIdx.x;
  if (t < (int)OFF_CVAL) ws[t] = 0u;
}

// ---------------- K1: streaming radix histogram of objectness ----------------
__global__ void k1_hist(const float* __restrict__ obj, unsigned* __restrict__ hist) {
  __shared__ unsigned lh[NB * NBIN];
  for (int t = threadIdx.x; t < NB * NBIN; t += blockDim.x) lh[t] = 0u;
  __syncthreads();
  int gid = blockIdx.x * blockDim.x + threadIdx.x;
  int gsz = gridDim.x * blockDim.x;
  for (int e = gid; e < TOTE; e += gsz) {
    __builtin_prefetch(obj + e + gsz, 0, 1);           // global_prefetch_b8
    float x = obj[e];
    int n = e / (AA * SP);
    unsigned b = fkey(x) >> 21;                        // 11-bit bin
    atomicAdd(&lh[n * NBIN + b], 1u);
  }
  __syncthreads();
  for (int t = threadIdx.x; t < NB * NBIN; t += blockDim.x)
    if (lh[t]) atomicAdd(&hist[t], lh[t]);
}

// ---------------- K2: pick threshold bin (suffix count >= PRE) ----------------
__global__ void k2_thr(const unsigned* __restrict__ hist, unsigned* __restrict__ thr) {
  int n = blockIdx.x;
  if (threadIdx.x == 0) {
    unsigned c = 0; int b = NBIN - 1;
    for (; b >= 0; --b) { c += hist[n * NBIN + b]; if (c >= PRE) break; }
    if (b < 0) b = 0;
    thr[n] = (unsigned)b;
  }
}

// ---------------- K3: compact candidates above threshold ----------------
__global__ void k3_compact(const float* __restrict__ obj, const unsigned* __restrict__ thr,
                           unsigned* __restrict__ cnt, float* __restrict__ cval,
                           unsigned* __restrict__ cidx) {
  int gid = blockIdx.x * blockDim.x + threadIdx.x;
  int gsz = gridDim.x * blockDim.x;
  for (int e = gid; e < TOTE; e += gsz) {
    float x = obj[e];
    int n = e / (AA * SP);
    unsigned b = fkey(x) >> 21;
    if (b >= thr[n]) {
      unsigned pos = atomicAdd(&cnt[n], 1u);
      if (pos < CAP) {
        int r = e - n * AA * SP;
        int a = r / SP, s = r - a * SP;
        cval[(size_t)n * CAP + pos] = x;
        cidx[(size_t)n * CAP + pos] = (unsigned)(s * AA + a); // transposed flat index
      }
    }
  }
}

// ---------------- K4: sort candidates, emit exact top-PRE ----------------
__global__ void k4_sort(const float* __restrict__ cval, const unsigned* __restrict__ cidx,
                        const unsigned* __restrict__ cnt, unsigned* __restrict__ tidx,
                        float* __restrict__ traw) {
  __shared__ unsigned long long skey[CAP];             // 64 KB LDS (WGP has 320 KB)
  int n = blockIdx.x;
  unsigned c = cnt[n]; if (c > CAP) c = CAP;
  for (int t = threadIdx.x; t < CAP; t += blockDim.x) {
    unsigned long long k = 0ull;
    if ((unsigned)t < c) {
      unsigned mk = fkey(cval[(size_t)n * CAP + t]);
      unsigned id = cidx[(size_t)n * CAP + t];
      k = (((unsigned long long)mk) << 32) | (unsigned long long)(0xFFFFFFFFu - id);
    }
    skey[t] = k;
  }
  __syncthreads();
  bitonic_desc(skey, CAP);
  for (int t = threadIdx.x; t < PRE; t += blockDim.x) {
    unsigned long long e = skey[t];
    unsigned id = 0u; float v = NEGV;
    if (e != 0ull) { id = 0xFFFFFFFFu - (unsigned)(e & 0xFFFFFFFFull); v = ikey((unsigned)(e >> 32)); }
    tidx[(size_t)n * PRE + t] = id;
    traw[(size_t)n * PRE + t] = v;
  }
}

// ---------------- K5: gather + decode + clip + validity + sigmoid ----------------
__global__ void k5_decode(const float* __restrict__ reg, const float* __restrict__ anc,
                          const unsigned* __restrict__ tidx, const float* __restrict__ traw,
                          float* __restrict__ props, unsigned* __restrict__ valid,
                          float* __restrict__ tsc) {
  int t = blockIdx.x * blockDim.x + threadIdx.x;
  if (t >= NB * PRE) return;
  int n = t / PRE;
  unsigned f = tidx[t];
  int a = (int)(f % AA);
  unsigned s = f / AA;
  float rg[6], an[6];
#pragma unroll
  for (int c = 0; c < 6; ++c) rg[c] = reg[((size_t)((n * AA + a) * 6 + c)) * SP + s];
#pragma unroll
  for (int c = 0; c < 6; ++c) an[c] = anc[((size_t)n * FLATN + f) * 6 + c];
  const float mx[3] = {255.f, 511.f, 511.f};
  float lo[3], hi[3]; bool ok = true;
#pragma unroll
  for (int d = 0; d < 3; ++d) {
    float sz = an[3 + d] - an[d] + 1.f;
    float ct = an[d] + 0.5f * sz;
    float dsz = fminf(rg[3 + d], CLIPV);
    float pc = rg[d] * sz + ct;
    float ps = expf(dsz) * sz;
    float l = pc - 0.5f * ps, h = pc + 0.5f * ps - 1.f;
    l = fminf(fmaxf(l, 0.f), mx[d]);
    h = fminf(fmaxf(h, 0.f), mx[d]);
    lo[d] = l; hi[d] = h;
    ok = ok && (h > l);
  }
  float* p = props + (size_t)t * 6;
#pragma unroll
  for (int d = 0; d < 3; ++d) { p[d] = lo[d]; p[3 + d] = hi[d]; }
  valid[t] = ok ? 1u : 0u;
  float sc = 1.f / (1.f + expf(-traw[t]));
  tsc[t] = ok ? sc : NEGV;
}

// ---------------- K6: suppression mask tiles; WMMA computes pairwise vol sums ----
// Branchless inner loop: iou > 0.7  <=>  inter > 0.7*union (union > 0 for all
// valid-valid pairs; pairs touching invalid boxes cannot influence the NMS).
__global__ __launch_bounds__(32) void k6_mask(const float* __restrict__ props,
                                              unsigned* __restrict__ mask) {
  const int lane = threadIdx.x;
  const int bx = blockIdx.x;            // 32-col word
  const int r0 = blockIdx.y * 16;       // 16-row tile
  const int n  = blockIdx.z;
  __shared__ float rb[16][6];
  __shared__ float rvol[16];
  if (lane < 16) {
    const float* p = props + ((size_t)n * PRE + (r0 + lane)) * 6;
    float v = 1.f;
#pragma unroll
    for (int c = 0; c < 6; ++c) rb[lane][c] = p[c];
#pragma unroll
    for (int d = 0; d < 3; ++d) v *= (rb[lane][3 + d] - rb[lane][d] + 1.f);
    rvol[lane] = v;
  }
  __syncthreads();

  const int nc = lane & 15;
  const int cA = bx * 32 + nc, cB = cA + 16;
  const bool inA = cA < PRE, inB = cB < PRE;
  float cbA[6], cbB[6];
  {
    const float* pA = props + ((size_t)n * PRE + cA) * 6;
    const float* pB = props + ((size_t)n * PRE + cB) * 6;
#pragma unroll
    for (int c = 0; c < 6; ++c) {
      cbA[c] = inA ? pA[c] : (c < 3 ? 0.f : -1.f);
      cbB[c] = inB ? pB[c] : (c < 3 ? 0.f : -1.f);
    }
  }
  float volA = 1.f, volB = 1.f;
#pragma unroll
  for (int d = 0; d < 3; ++d) {
    volA *= (cbA[3 + d] - cbA[d] + 1.f);
    volB *= (cbB[3 + d] - cbB[d] + 1.f);
  }
  if (!inA) volA = 0.f;
  if (!inB) volB = 0.f;

  // A matrix: row m = (hi(vol_m), lo(vol_m), 1, 1, 0...)  (lanes 0-15 hold K=0..7 halves)
  v16h Ah = {};
  {
    float vr = (lane < 16 ? rvol[lane] : 0.f) * VSCALE;
    _Float16 h = (_Float16)vr;
    _Float16 l = (_Float16)(vr - (float)h);
    Ah[0] = (lane < 16) ? h : (_Float16)0.f;
    Ah[1] = (lane < 16) ? l : (_Float16)0.f;
    Ah[2] = (lane < 16) ? (_Float16)1.f : (_Float16)0.f;
    Ah[3] = (lane < 16) ? (_Float16)1.f : (_Float16)0.f;
  }
  // B matrices: col j = (1, 1, hi(vol_j), lo(vol_j), 0...) (lanes 0-15 hold K=0..7)
  v16h BhA = {}, BhB = {};
  {
    float va = volA * VSCALE, vb = volB * VSCALE;
    _Float16 ha = (_Float16)va; _Float16 la = (_Float16)(va - (float)ha);
    _Float16 hb = (_Float16)vb; _Float16 lb = (_Float16)(vb - (float)hb);
    bool low = lane < 16;
    BhA[0] = low ? (_Float16)1.f : (_Float16)0.f;
    BhA[1] = low ? (_Float16)1.f : (_Float16)0.f;
    BhA[2] = low ? ha : (_Float16)0.f;
    BhA[3] = low ? la : (_Float16)0.f;
    BhB[0] = low ? (_Float16)1.f : (_Float16)0.f;
    BhB[1] = low ? (_Float16)1.f : (_Float16)0.f;
    BhB[2] = low ? hb : (_Float16)0.f;
    BhB[3] = low ? lb : (_Float16)0.f;
  }
  v8f accA = {}, accB = {};
  accA = __builtin_amdgcn_wmma_f32_16x16x32_f16(false, Ah, false, BhA, (short)0, accA, false, false);
  accB = __builtin_amdgcn_wmma_f32_16x16x32_f16(false, Ah, false, BhB, (short)0, accB, false, false);

  const int Mbase = (lane >> 4) << 3;   // C/D layout: lanes 16-31 hold rows 8..15
#pragma unroll
  for (int k = 0; k < 8; ++k) {
    const int M = k + Mbase;
    const int gi = r0 + M;
    float interA = 1.f, interB = 1.f;
#pragma unroll
    for (int d = 0; d < 3; ++d) {
      float l1 = rb[M][d], h1 = rb[M][3 + d];
      interA *= fmaxf(fminf(h1, cbA[3 + d]) - fmaxf(l1, cbA[d]) + 1.f, 0.f);
      interB *= fmaxf(fminf(h1, cbB[3 + d]) - fmaxf(l1, cbB[d]) + 1.f, 0.f);
    }
    float unA = accA[k] * VSCALI - interA;   // vol_i + vol_j from the matrix core
    float unB = accB[k] * VSCALI - interB;
    // branchless: bitwise & of plain compares (no exec-mask save/restore, no fdiv)
    bool pA = (interA > 0.7f * unA) & inA & (cA > gi);
    bool pB = (interB > 0.7f * unB) & inB & (cB > gi);
    unsigned wAk = ballot32(pA);
    unsigned wBk = ballot32(pB);
    // rows k (lanes 0-15) and k+8 (lanes 16-31), 32 cols per word
    unsigned wrow  = (wAk & 0xFFFFu) | ((wBk & 0xFFFFu) << 16);
    unsigned wrow8 = (wAk >> 16)     | (wBk & 0xFFFF0000u);
    if (lane == 0) {
      mask[((size_t)n * PRE + (r0 + k)) * WORDS + bx] = wrow;
      mask[((size_t)n * PRE + (r0 + k + 8)) * WORDS + bx] = wrow8;
    }
  }
#if __has_builtin(__builtin_amdgcn_s_wait_tensorcnt)
  __builtin_amdgcn_s_wait_tensorcnt(0);
#endif
}

// ---------------- K7: sequential greedy suppression over bitmask ----------------
__global__ void k7_nms(const unsigned* __restrict__ valid, const unsigned* __restrict__ mask,
                       const float* __restrict__ tsc, float* __restrict__ sc2) {
  __shared__ unsigned keepw[WORDS];
  int n = blockIdx.x;
  int tid = threadIdx.x;
  if (tid < WORDS) {
    unsigned w = 0;
    for (int b = 0; b < 32; ++b) {
      int j = tid * 32 + b;
      if (j < PRE && valid[(size_t)n * PRE + j]) w |= (1u << b);
    }
    keepw[tid] = w;
  }
  __syncthreads();
  for (int i = 0; i < PRE; ++i) {
    unsigned kw = keepw[i >> 5];
    if ((kw >> (i & 31)) & 1u) {
      if (tid < WORDS)
        keepw[tid] &= ~mask[((size_t)n * PRE + i) * WORDS + tid];
    }
    __syncthreads();
  }
  for (int j = tid; j < PRE; j += blockDim.x) {
    bool kept = (keepw[j >> 5] >> (j & 31)) & 1u;
    sc2[(size_t)n * PRE + j] = kept ? tsc[(size_t)n * PRE + j] : NEGV;
  }
}

// ---------------- K8: final top-POST, write boxes + scores ----------------
__global__ void k8_final(const float* __restrict__ sc2, const float* __restrict__ props,
                         float* __restrict__ out) {
  __shared__ unsigned long long skey[2048];
  int n = blockIdx.x;
  for (int t = threadIdx.x; t < 2048; t += blockDim.x) {
    unsigned long long k = 0ull;
    if (t < PRE) {
      unsigned mk = fkey(sc2[(size_t)n * PRE + t]);
      k = (((unsigned long long)mk) << 32) | (unsigned long long)(0xFFFFFFFFu - (unsigned)t);
    }
    skey[t] = k;
  }
  __syncthreads();
  bitonic_desc(skey, 2048);
  float* ob = out;                        // (NB, POST, 6)
  float* os = out + (size_t)NB * POST * 6;// (NB, POST)
  for (int t = threadIdx.x; t < POST; t += blockDim.x) {
    unsigned long long e = skey[t];
    unsigned j = 0u; float sc = NEGV;
    if (e != 0ull) { j = 0xFFFFFFFFu - (unsigned)(e & 0xFFFFFFFFull); sc = ikey((unsigned)(e >> 32)); }
    os[(size_t)n * POST + t] = sc;
    const float* p = props + ((size_t)n * PRE + j) * 6;
#pragma unroll
    for (int c = 0; c < 6; ++c) ob[((size_t)n * POST + t) * 6 + c] = p[c];
  }
}

// ---------------- launch ----------------
extern "C" void kernel_launch(void* const* d_in, const int* in_sizes, int n_in,
                              void* d_out, int out_size, void* d_ws, size_t ws_size,
                              hipStream_t stream) {
  (void)in_sizes; (void)n_in; (void)out_size; (void)ws_size;
  const float* obj = (const float*)d_in[0];
  const float* reg = (const float*)d_in[1];
  const float* anc = (const float*)d_in[2];
  float* out = (float*)d_out;
  unsigned* ws = (unsigned*)d_ws;

  unsigned* hist = ws + OFF_HIST;
  unsigned* cnt  = ws + OFF_CNT;
  unsigned* thr  = ws + OFF_THR;
  float*    cval = (float*)(ws + OFF_CVAL);
  unsigned* cidx = ws + OFF_CIDX;
  unsigned* tidx = ws + OFF_TIDX;
  float*    traw = (float*)(ws + OFF_TRAW);
  float*    tsc  = (float*)(ws + OFF_TSC);
  float*    props= (float*)(ws + OFF_PROP);
  unsigned* valid= ws + OFF_VAL;
  float*    sc2  = (float*)(ws + OFF_SC2);
  unsigned* mask = ws + OFF_MASK;

  k0_zero<<<(int)((OFF_CVAL + 255) / 256), 256, 0, stream>>>(ws);
  k1_hist<<<3072, 256, 0, stream>>>(obj, hist);
  k2_thr<<<NB, 32, 0, stream>>>(hist, thr);
  k3_compact<<<3072, 256, 0, stream>>>(obj, thr, cnt, cval, cidx);
  k4_sort<<<NB, 1024, 0, stream>>>(cval, cidx, cnt, tidx, traw);
  k5_decode<<<(NB * PRE + 255) / 256, 256, 0, stream>>>(reg, anc, tidx, traw, props, valid, tsc);
  k6_mask<<<dim3(CTILES, RTILES, NB), 32, 0, stream>>>(props, mask);
  k7_nms<<<NB, 64, 0, stream>>>(valid, mask, tsc, sc2);
  k8_final<<<NB, 1024, 0, stream>>>(sc2, props, out);
}
